// Double_SSM_Block_Encoder_59382217834750
// MI455X (gfx1250) — matrix operand order repaired and checked by reading, not compiled, not run on hardware
//
#include <hip/hip_runtime.h>
#include <hip/hip_bf16.h>
#include <math.h>

typedef __attribute__((ext_vector_type(16))) __bf16 v16bf;
typedef __attribute__((ext_vector_type(8)))  float  v8f;

#define BATCH  8
#define SEQ    4096
#define ROWS   (BATCH*SEQ)   /* 32768 */
#define DIN    128
#define NSTATE 16
#define NDTR   4

__device__ __forceinline__ float sigmoidf_(float x){ return 1.f/(1.f+__expf(-x)); }
__device__ __forceinline__ float siluf_(float x){ return x*sigmoidf_(x); }
__device__ __forceinline__ float softplusf_(float x){ return (x > 20.f) ? x : log1pf(__expf(x)); }

// ---------------------------------------------------------------------------
// GEMM: Out[M,N] = A[M,K] @ W[K,N], fp32 in/out, bf16-split WMMA (3 wmma per
// 32-wide K chunk -> ~fp32 accuracy; error term a_lo*w_lo ~ 2^-16 relative).
// One wave (wave32) per 16x16 tile; grids sized exactly so EXEC is all ones
// for every WMMA (ISA 7.12 requirement).
// TRANSA: A is the input tensor x in (b, c, h*w) layout:
//         A[m,k] = x[(b*64 + k)*4096 + l],  b = m>>12, l = m&4095.
// K and TRANSA are compile-time so the K loop fully unrolls and the address
// selection has no per-element branches.
// ---------------------------------------------------------------------------
template<int K, bool TRANSA>
__global__ __launch_bounds__(256) void gemm_bf16split_wmma(
    const float* __restrict__ A, const float* __restrict__ W,
    float* __restrict__ Out, int N)
{
  const int wave  = blockIdx.x * (blockDim.x >> 5) + (threadIdx.x >> 5);
  const int lane  = threadIdx.x & 31;
  const int tilesN = N >> 4;
  const int tM = wave / tilesN;
  const int tN = wave - tM * tilesN;

  const int laneLo = lane & 15;
  const int laneHi = lane >> 4;
  const int mrow = tM*16 + laneLo;   // A row for this lane's fragment elements
  const int ncol = tN*16 + laneLo;   // B column for this lane's fragment elements

  // Precomputed per-lane A base (element loop adds only the K index).
  const float* Abase;
  if (TRANSA) {
    const int bb = mrow >> 12, ll = mrow & 4095;
    Abase = A + (size_t)bb * 64 * 4096 + ll;     // + ka*4096
  } else {
    Abase = A + (size_t)mrow * K;                // + ka
  }
  const float* Wbase = W + ncol;                 // + kw*N

  v8f acc = {};
  #pragma unroll
  for (int kb = 0; kb < K; kb += 32) {
    v16bf ah, al, wh, wl;
    #pragma unroll
    for (int e = 0; e < 16; ++e) {
      // A 16x32 bf16 fragment: K = kb + (e/8)*16 + (lane/16)*8 + (e%8)
      const int ka = kb + ((e >> 3) << 4) + (laneHi << 3) + (e & 7);
      const float av = TRANSA ? Abase[(size_t)ka * 4096] : Abase[ka];
      const __bf16 avh = (__bf16)av;
      ah[e] = avh;
      al[e] = (__bf16)(av - (float)avh);

      // B 32x16 bf16 fragment: K = kb + (e/8)*16 + (lane/16)*8 + (e%8)
      const int kw = kb + ((e >> 3) << 4) + (laneHi << 3) + (e & 7);
      const float wv = Wbase[(size_t)kw * N];
      const __bf16 wvh = (__bf16)wv;
      wh[e] = wvh;
      wl[e] = (__bf16)(wv - (float)wvh);
    }
    acc = __builtin_amdgcn_wmma_f32_16x16x32_bf16(false, ah, false, wh, (short)0, acc, false, false);
    acc = __builtin_amdgcn_wmma_f32_16x16x32_bf16(false, ah, false, wl, (short)0, acc, false, false);
    acc = __builtin_amdgcn_wmma_f32_16x16x32_bf16(false, al, false, wh, (short)0, acc, false, false);
  }
  // D 16x16 f32: VGPR v holds M = v + 8*(lane/16), N = lane%16
  #pragma unroll
  for (int v = 0; v < 8; ++v) {
    const int m = tM*16 + v + (laneHi << 3);
    Out[(size_t)m * N + ncol] = acc[v];
  }
}

// ---------------------------------------------------------------------------
// Depthwise causal conv (kernel 4, left pad 3) + SiLU.  xc = silu(conv(xz[:, :128]))
// ---------------------------------------------------------------------------
__global__ __launch_bounds__(256) void conv_silu_kernel(
    const float* __restrict__ xz, const float* __restrict__ w_conv,
    const float* __restrict__ b_conv, float* __restrict__ xc)
{
  const int idx = blockIdx.x * blockDim.x + threadIdx.x;   // ROWS*DIN
  const int d   = idx & 127;
  const int row = idx >> 7;            // b*SEQ + l
  const int l   = row & 4095;
  float s = b_conv[d];
  #pragma unroll
  for (int k = 0; k < 4; ++k) {
    const int ls = l + k - 3;
    if (ls >= 0) s += w_conv[d*4 + k] * xz[(size_t)(row + k - 3) * 256 + d];
  }
  xc[idx] = siluf_(s);
}

// ---------------------------------------------------------------------------
// proj = xc @ w_x (128x36); dt = softplus(proj[:4] @ w_dt + b_dt); B,C splits.
// One thread per (b,l) row.
// ---------------------------------------------------------------------------
__global__ __launch_bounds__(256) void proj_kernel(
    const float* __restrict__ xc, const float* __restrict__ w_x,
    const float* __restrict__ w_dt, const float* __restrict__ b_dt,
    float* __restrict__ dt_out, float* __restrict__ B_out, float* __restrict__ C_out)
{
  const int row = blockIdx.x * blockDim.x + threadIdx.x;
  const float* xr = xc + (size_t)row * DIN;
  float proj[NDTR + 2*NSTATE];
  #pragma unroll
  for (int j = 0; j < NDTR + 2*NSTATE; ++j) proj[j] = 0.f;
  for (int dd = 0; dd < DIN; ++dd) {
    const float xv = xr[dd];
    const float* wr = w_x + dd * (NDTR + 2*NSTATE);
    #pragma unroll
    for (int j = 0; j < NDTR + 2*NSTATE; ++j) proj[j] += xv * wr[j];
  }
  #pragma unroll
  for (int nn = 0; nn < NSTATE; ++nn) {
    B_out[(size_t)row*NSTATE + nn] = proj[NDTR + nn];
    C_out[(size_t)row*NSTATE + nn] = proj[NDTR + NSTATE + nn];
  }
  for (int dd = 0; dd < DIN; ++dd) {
    float a = b_dt[dd];
    #pragma unroll
    for (int r = 0; r < NDTR; ++r) a += proj[r] * w_dt[r*DIN + dd];
    dt_out[(size_t)row*DIN + dd] = softplusf_(a);
  }
}

// ---------------------------------------------------------------------------
// Selective scan. 16 lanes per (b,d) recurrence (one state per lane), 1024
// recurrences. h = exp(dt*A)*h + dt*xc*B; y = sum_n h*C via shfl_xor butterfly.
// Prefetches (global_prefetch_b8) hide load latency on the serial L chain.
// ---------------------------------------------------------------------------
__global__ __launch_bounds__(256) void scan_kernel(
    const float* __restrict__ dt, const float* __restrict__ xc,
    const float* __restrict__ Bm, const float* __restrict__ Cm,
    const float* __restrict__ A_log, float* __restrict__ Y)
{
  const int tid = blockIdx.x * blockDim.x + threadIdx.x;
  const int grp = tid >> 4;            // (b,d): 0..1023
  const int n   = tid & 15;
  const int b   = grp >> 7;
  const int d   = grp & 127;
  const float Adn = -__expf(A_log[d*NSTATE + n]);
  float h = 0.f;
  const size_t base = (size_t)b * SEQ;
  for (int t = 0; t < SEQ; ++t) {
    const size_t row = base + t;
    if (t + 8 < SEQ) {
      __builtin_prefetch(&Bm[(row + 8)*NSTATE + n], 0, 0);
      __builtin_prefetch(&Cm[(row + 8)*NSTATE + n], 0, 0);
      __builtin_prefetch(&dt[(row + 8)*DIN + d], 0, 0);
    }
    const float dtv = dt[row*DIN + d];
    const float xv  = xc[row*DIN + d];
    const float bv  = Bm[row*NSTATE + n];
    const float cv  = Cm[row*NSTATE + n];
    h = __expf(dtv * Adn) * h + dtv * xv * bv;
    float p = h * cv;
    p += __shfl_xor(p, 1, 32);
    p += __shfl_xor(p, 2, 32);
    p += __shfl_xor(p, 4, 32);
    p += __shfl_xor(p, 8, 32);
    if (n == 0) Y[row*DIN + d] = p;
  }
}

// ---------------------------------------------------------------------------
// y = (y_scan + xc*D) * silu(z)   (z = xz[:, 128:256]); in-place on Y.
// ---------------------------------------------------------------------------
__global__ __launch_bounds__(256) void gate_kernel(
    float* __restrict__ Y, const float* __restrict__ xc,
    const float* __restrict__ xz, const float* __restrict__ Dskip)
{
  const int idx = blockIdx.x * blockDim.x + threadIdx.x;
  const int d   = idx & 127;
  const int row = idx >> 7;
  const float z = xz[(size_t)row*256 + 128 + d];
  Y[idx] = (Y[idx] + xc[idx]*Dskip[d]) * siluf_(z);
}

// ---------------------------------------------------------------------------
// LayerNorm (eps 1e-5) + ReLU. PERMUTE fuses the final (0,3,2,1) transpose:
// out[((b*128 + c)*64 + w)*64 + h], l = h*64 + w.
// ---------------------------------------------------------------------------
template<int N, bool PERMUTE>
__global__ __launch_bounds__(256) void ln_relu_kernel(
    const float* __restrict__ in, float* __restrict__ out,
    const float* __restrict__ g, const float* __restrict__ bln)
{
  const int row = blockIdx.x * blockDim.x + threadIdx.x;
  const float* xr = in + (size_t)row * N;
  float mu = 0.f;
  #pragma unroll 8
  for (int j = 0; j < N; ++j) mu += xr[j];
  mu /= (float)N;
  float var = 0.f;
  #pragma unroll 8
  for (int j = 0; j < N; ++j) { const float df = xr[j] - mu; var += df*df; }
  var /= (float)N;
  const float inv = rsqrtf(var + 1e-5f);
  if (PERMUTE) {
    const int b = row >> 12, l = row & 4095;
    const int hh = l >> 6, ww = l & 63;
    for (int j = 0; j < N; ++j) {
      const float v = (xr[j]-mu)*inv*g[j] + bln[j];
      out[(((size_t)b*128 + j)*64 + ww)*64 + hh] = fmaxf(v, 0.f);
    }
  } else {
    for (int j = 0; j < N; ++j) {
      const float v = (xr[j]-mu)*inv*g[j] + bln[j];
      out[(size_t)row*N + j] = fmaxf(v, 0.f);
    }
  }
}

extern "C" void kernel_launch(void* const* d_in, const int* in_sizes, int n_in,
                              void* d_out, int out_size, void* d_ws, size_t ws_size,
                              hipStream_t stream)
{
  const float* x      = (const float*)d_in[0];
  const float* w_in1  = (const float*)d_in[1];
  const float* w_conv1= (const float*)d_in[2];
  const float* b_conv1= (const float*)d_in[3];
  const float* w_x1   = (const float*)d_in[4];
  const float* w_dt1  = (const float*)d_in[5];
  const float* b_dt1  = (const float*)d_in[6];
  const float* A_log1 = (const float*)d_in[7];
  const float* D1     = (const float*)d_in[8];
  const float* w_out1 = (const float*)d_in[9];
  const float* g_ln1  = (const float*)d_in[10];
  const float* b_ln1  = (const float*)d_in[11];
  const float* w_in2  = (const float*)d_in[12];
  const float* w_conv2= (const float*)d_in[13];
  const float* b_conv2= (const float*)d_in[14];
  const float* w_x2   = (const float*)d_in[15];
  const float* w_dt2  = (const float*)d_in[16];
  const float* b_dt2  = (const float*)d_in[17];
  const float* A_log2 = (const float*)d_in[18];
  const float* D2     = (const float*)d_in[19];
  const float* w_out2 = (const float*)d_in[20];
  const float* g_ln2  = (const float*)d_in[21];
  const float* b_ln2  = (const float*)d_in[22];

  // workspace layout (floats); Ob aliases xz (z half consumed before out-proj)
  float* ws  = (float*)d_ws;
  float* t1  = ws;                          // ROWS*64   block-1 output
  float* xz  = t1  + (size_t)ROWS*64;       // ROWS*256
  float* xcb = xz  + (size_t)ROWS*256;      // ROWS*128
  float* dtb = xcb + (size_t)ROWS*128;      // ROWS*128
  float* Bb  = dtb + (size_t)ROWS*128;      // ROWS*16
  float* Cb  = Bb  + (size_t)ROWS*16;       // ROWS*16
  float* Yb  = Cb  + (size_t)ROWS*16;       // ROWS*128
  float* Ob  = xz;                          // out-proj result reuses xz

  const int ELEMS = ROWS * DIN;

  // ================= block 1 =================
  gemm_bf16split_wmma<64, true ><<<((ROWS/16)*(256/16))/8, 256, 0, stream>>>(x,  w_in1, xz, 256);
  conv_silu_kernel<<<ELEMS/256, 256, 0, stream>>>(xz, w_conv1, b_conv1, xcb);
  proj_kernel<<<ROWS/256, 256, 0, stream>>>(xcb, w_x1, w_dt1, b_dt1, dtb, Bb, Cb);
  scan_kernel<<<(BATCH*DIN*NSTATE)/256, 256, 0, stream>>>(dtb, xcb, Bb, Cb, A_log1, Yb);
  gate_kernel<<<ELEMS/256, 256, 0, stream>>>(Yb, xcb, xz, D1);
  gemm_bf16split_wmma<128, false><<<((ROWS/16)*(64/16))/8, 256, 0, stream>>>(Yb, w_out1, Ob, 64);
  ln_relu_kernel<64, false><<<ROWS/256, 256, 0, stream>>>(Ob, t1, g_ln1, b_ln1);

  // ================= block 2 =================
  gemm_bf16split_wmma<64, false><<<((ROWS/16)*(256/16))/8, 256, 0, stream>>>(t1, w_in2, xz, 256);
  conv_silu_kernel<<<ELEMS/256, 256, 0, stream>>>(xz, w_conv2, b_conv2, xcb);
  proj_kernel<<<ROWS/256, 256, 0, stream>>>(xcb, w_x2, w_dt2, b_dt2, dtb, Bb, Cb);
  scan_kernel<<<(BATCH*DIN*NSTATE)/256, 256, 0, stream>>>(dtb, xcb, Bb, Cb, A_log2, Yb);
  gate_kernel<<<ELEMS/256, 256, 0, stream>>>(Yb, xcb, xz, D2);
  gemm_bf16split_wmma<128, false><<<((ROWS/16)*(128/16))/8, 256, 0, stream>>>(Yb, w_out2, Ob, 128);
  ln_relu_kernel<128, true><<<ROWS/256, 256, 0, stream>>>(Ob, (float*)d_out, g_ln2, b_ln2);

  (void)in_sizes; (void)n_in; (void)out_size; (void)ws_size;
}
